// MentionScore_910533066864
// MI455X (gfx1250) — matrix operand order repaired and checked by reading, not compiled, not run on hardware
//
#include <hip/hip_runtime.h>
#include <hip/hip_bf16.h>
#include <cstddef>

// ---------------- problem constants (from reference) ----------------
#define T_TOK 50000
#define S_SPAN 100000
#define DS 400     // state dim
#define DE 350     // embed dim
#define HH 150     // hidden
#define DW 20      // width-emb dim
#define WMAXX 10   // max span width

// WMMA tiling (16x16x32 bf16)
#define NT 10      // ceil(150/16) N-tiles
#define LDH 160    // padded hidden row stride (= 5*32)
#define LDP 352    // padded pooled row stride (= 11*32)
#define KT_DS 13   // ceil(400/32): 12 full + 1 partial
#define KT_DE 11   // 352/32 full tiles (pad zeroed)
#define KT_H 5     // 160/32 full tiles

// sw1 packed K-segment tile bases: [x_start | x_end | pooled | width_emb]
#define SEG0_BASE 0
#define SEG1_BASE 13
#define SEG2_BASE 26
#define SEG3_BASE 37
#define SW1_TILES 38

typedef __attribute__((ext_vector_type(16))) __bf16 v16bf;
typedef __attribute__((ext_vector_type(8)))  __bf16 v8bf;
typedef __attribute__((ext_vector_type(8)))  float  v8f;
typedef __attribute__((ext_vector_type(4)))  float  fl4;

__device__ __forceinline__ v8f vzero8() {
  v8f z;
#pragma unroll
  for (int i = 0; i < 8; ++i) z[i] = 0.0f;
  return z;
}

__device__ __forceinline__ v8f wmma_bf16(v16bf a, v16bf b, v8f c) {
  return __builtin_amdgcn_wmma_f32_16x16x32_bf16(false, a, false, b, (short)0, c,
                                                 false, false);
}

// ---- A fragment, full k-tile, from per-lane global fp32 row (16B aligned) ----
// A layout: lane L holds row M=L%16; element e -> k = k0 + e + (e>=8?8:0) + half*8
__device__ __forceinline__ v16bf a_frag_full(const float* __restrict__ row, int k0,
                                             int half) {
  const fl4* p = (const fl4*)(row + k0 + half * 8);
  fl4 x0 = p[0], x1 = p[1], x2 = p[4], x3 = p[5];  // k0+h8+[0..7], k0+h8+16+[0..7]
  v16bf a;
#pragma unroll
  for (int j = 0; j < 4; ++j) {
    a[j]      = (__bf16)x0[j];
    a[4 + j]  = (__bf16)x1[j];
    a[8 + j]  = (__bf16)x2[j];
    a[12 + j] = (__bf16)x3[j];
  }
  return a;
}

// ---- A fragment, partial k-tile: clamp index (padded B rows are zero, so the
// clamped-duplicate values contribute nothing; unconditional in-bounds loads) ----
__device__ __forceinline__ v16bf a_frag_partial(const float* __restrict__ row, int k0,
                                                int kmax, int half) {
  v16bf a;
#pragma unroll
  for (int e = 0; e < 16; ++e) {
    int k = k0 + e + ((e >= 8) ? 8 : 0) + (half << 3);
    k = (k > kmax) ? kmax : k;  // v_min, no branch
    a[e] = (__bf16)row[k];
  }
  return a;
}

// ---- A fragment from bf16 LDS row (fully padded, guard-free, 16B runs) ----
__device__ __forceinline__ v16bf a_frag_lds(const __bf16* row, int k0, int half) {
  const v8bf* p = (const v8bf*)(row + k0 + half * 8);
  v8bf lo = p[0], hi = p[2];
  v16bf a;
#pragma unroll
  for (int j = 0; j < 8; ++j) {
    a[j] = lo[j];
    a[8 + j] = hi[j];
  }
  return a;
}

// ---- B fragment from pre-packed fragment-order bf16: 32 contiguous bytes ----
__device__ __forceinline__ v16bf b_frag_packed(const __bf16* __restrict__ pack,
                                               int tile, int lane) {
  const v8bf* p = (const v8bf*)(pack + ((size_t)tile * 32 + lane) * 16);
  v8bf lo = p[0], hi = p[1];
  v16bf b;
#pragma unroll
  for (int j = 0; j < 8; ++j) {
    b[j] = lo[j];
    b[8 + j] = hi[j];
  }
  return b;
}

// ---- store accumulator tile to LDS h with bias + ReLU (branch-free) ----
// C/D layout: VGPR r holds M = r + half*8, N = lane%16.
__device__ __forceinline__ void store_h(__bf16* h, const v8f* acc,
                                        const float* __restrict__ bias, int lane) {
  int n16 = lane & 15, half = lane >> 4;
#pragma unroll
  for (int nt = 0; nt < NT; ++nt) {
    int n = nt * 16 + n16;
    int nb = (n < HH) ? n : (HH - 1);  // clamp; pad cols hold finite garbage (ok:
    float bv = bias[nb];               //  packed B rows >=150 are zero)
#pragma unroll
    for (int r = 0; r < 8; ++r) {
      int m = half * 8 + r;
      h[m * LDH + n] = (__bf16)fmaxf(acc[nt][r] + bv, 0.0f);
    }
  }
}

// ---- shared tail: ReLU(h@w2+b2) @ w3 + b3 -> out[rowbase+M] ----
__device__ __forceinline__ void mlp_tail(__bf16* h, const __bf16* __restrict__ w2p,
                                         const float* __restrict__ b2,
                                         const __bf16* __restrict__ w3p,
                                         const float* __restrict__ b3,
                                         float* __restrict__ out, int rowbase,
                                         int rowlimit, int lane) {
  int n16 = lane & 15, half = lane >> 4;
  __syncthreads();  // layer-1 h stores visible

  v8f acc[NT];
#pragma unroll
  for (int i = 0; i < NT; ++i) acc[i] = vzero8();
  for (int kt = 0; kt < KT_H; ++kt) {
    v16bf a = a_frag_lds(h + (size_t)n16 * LDH, kt * 32, half);
#pragma unroll
    for (int nt = 0; nt < NT; ++nt)
      acc[nt] = wmma_bf16(a, b_frag_packed(w2p, kt * NT + nt, lane), acc[nt]);
  }
  __syncthreads();            // all layer-2 reads of h done
  store_h(h, acc, b2, lane);  // overwrite h with layer-2 activations
  __syncthreads();

  v8f acc3 = vzero8();
  for (int kt = 0; kt < KT_H; ++kt) {
    v16bf a = a_frag_lds(h + (size_t)n16 * LDH, kt * 32, half);
    acc3 = wmma_bf16(a, b_frag_packed(w3p, kt, lane), acc3);
  }
  if (n16 == 0) {  // lanes 0 (rows 0..7) and 16 (rows 8..15) hold column 0
    float bv = b3[0];
#pragma unroll
    for (int r = 0; r < 8; ++r) {
      int row = rowbase + half * 8 + r;
      if (row < rowlimit) out[row] = acc3[r] + bv;
    }
  }
}

// ---------------- pack kernel: fp32 weights -> fragment-order padded bf16 ----
// dst[((kt*ntiles+nt)*32+lane)*16+e] = W[rowoff + kt*32+(lane>>4)*8+e+(e>=8?8:0)][nt*16+lane%16]
__global__ void pack_b_kernel(const float* __restrict__ W, int ldn, int rowoff,
                              int kvalid, int ntiles, int nvalid, int total,
                              __bf16* __restrict__ dst) {
  int idx = blockIdx.x * blockDim.x + threadIdx.x;
  if (idx >= total) return;
  int e = idx & 15;
  int lane = (idx >> 4) & 31;
  int tile = idx >> 9;
  int nt = tile % ntiles;
  int kt = tile / ntiles;
  int n = nt * 16 + (lane & 15);
  int kl = kt * 32 + (lane >> 4) * 8 + e + ((e >= 8) ? 8 : 0);
  float v = ((kl < kvalid) && (n < nvalid)) ? W[(size_t)(rowoff + kl) * ldn + n] : 0.0f;
  dst[idx] = (__bf16)v;
}

// ---------------- kernel 1: per-token attention logits ----------------
__global__ void __launch_bounds__(64) attn_logits_kernel(
    const float* __restrict__ states, const __bf16* __restrict__ w1p,
    const float* __restrict__ b1, const __bf16* __restrict__ w2p,
    const float* __restrict__ b2, const __bf16* __restrict__ w3p,
    const float* __restrict__ b3, float* __restrict__ attns) {
  __shared__ __bf16 hS[2][16 * LDH];
  int lane = threadIdx.x & 31, wave = threadIdx.x >> 5;
  int n16 = lane & 15, half = lane >> 4;

  int rowbase = (blockIdx.x * 2 + wave) * 16;
  int arow = rowbase + n16;
  if (arow > T_TOK - 1) arow = T_TOK - 1;  // clamped dup work; store guarded
  const float* xrow = states + (size_t)arow * DS;

  v8f acc[NT];
#pragma unroll
  for (int i = 0; i < NT; ++i) acc[i] = vzero8();
  for (int kt = 0; kt < KT_DS - 1; ++kt) {  // 12 full tiles
    v16bf a = a_frag_full(xrow, kt * 32, half);
#pragma unroll
    for (int nt = 0; nt < NT; ++nt)
      acc[nt] = wmma_bf16(a, b_frag_packed(w1p, kt * NT + nt, lane), acc[nt]);
  }
  {  // partial tile: k 384..415, clamp to 399 (packed B rows >=400 are zero)
    v16bf a = a_frag_partial(xrow, (KT_DS - 1) * 32, DS - 1, half);
#pragma unroll
    for (int nt = 0; nt < NT; ++nt)
      acc[nt] = wmma_bf16(a, b_frag_packed(w1p, (KT_DS - 1) * NT + nt, lane), acc[nt]);
  }
  store_h(hS[wave], acc, b1, lane);
  mlp_tail(hS[wave], w2p, b2, w3p, b3, attns, rowbase, T_TOK, lane);
}

// ---------------- kernel 2: fused span scorer ----------------
__global__ void __launch_bounds__(64) span_score_kernel(
    const float* __restrict__ states, const float* __restrict__ embeds,
    const int* __restrict__ sstart, const int* __restrict__ swidth,
    const float* __restrict__ wtab, const __bf16* __restrict__ w1p,
    const float* __restrict__ b1, const __bf16* __restrict__ w2p,
    const float* __restrict__ b2, const __bf16* __restrict__ w3p,
    const float* __restrict__ b3, const float* __restrict__ attns,
    float* __restrict__ out) {
  __shared__ float  pS[2][16][WMAXX];
  __shared__ int    stS[2][16];
  __shared__ __bf16 pooledS[2][16 * LDP];
  __shared__ __bf16 hS[2][16 * LDH];

  int lane = threadIdx.x & 31, wave = threadIdx.x >> 5;
  int n16 = lane & 15, half = lane >> 4;
  int spanbase = (blockIdx.x * 2 + wave) * 16;
  int sp = spanbase + n16;
  if (sp > S_SPAN - 1) sp = S_SPAN - 1;

  int st = sstart[sp];
  int wd = swidth[sp];
  int en = st + wd - 1;
  if (en > T_TOK - 1) en = T_TOK - 1;
  int bin = (wd >= 1) + (wd >= 2) + (wd >= 3) + (wd >= 4) + (wd >= 8) + (wd >= 16) +
            (wd >= 32) + (wd >= 64);

  // ---- masked softmax over the span window (lanes 0..15, one span each) ----
  if (lane < 16) {
    float m = -3.0e38f;
    for (int w = 0; w < wd && w < WMAXX; ++w) {
      int ix = st + w;
      if (ix > T_TOK - 1) ix = T_TOK - 1;
      m = fmaxf(m, attns[ix]);
    }
    float ex[WMAXX];
    float den = 0.0f;
#pragma unroll
    for (int w = 0; w < WMAXX; ++w) {
      float e = 0.0f;
      if (w < wd) {
        int ix = st + w;
        if (ix > T_TOK - 1) ix = T_TOK - 1;
        e = __expf(attns[ix] - m);
      }
      ex[w] = e;
      den += e;
    }
    float inv = 1.0f / den;
#pragma unroll
    for (int w = 0; w < WMAXX; ++w) pS[wave][lane][w] = ex[w] * inv;
    stS[wave][lane] = st;
  }
  __syncthreads();

  // ---- attention-weighted pooling of embeds -> LDS (bf16) ----
  for (int d = lane; d < DE; d += 32) {
    for (int s = 0; s < 16; ++s) {
      int s0 = stS[wave][s];
      float accp = 0.0f;
#pragma unroll
      for (int w = 0; w < WMAXX; ++w) {
        float pw = pS[wave][s][w];
        if (pw != 0.0f) {
          int ix = s0 + w;
          if (ix > T_TOK - 1) ix = T_TOK - 1;
          accp += pw * embeds[(size_t)ix * DE + d];
        }
      }
      pooledS[wave][s * LDP + d] = (__bf16)accp;
    }
  }
  // zero the 2 pad columns (350,351): 32 lanes cover 16 rows x 2 cols exactly
  {
    int s = lane >> 1, c = DE + (lane & 1);
    pooledS[wave][s * LDP + c] = (__bf16)0.0f;
  }
  __syncthreads();

  // ---- layer 1: g @ sw1 without materializing g; 4 packed K-segments ----
  v8f acc[NT];
#pragma unroll
  for (int i = 0; i < NT; ++i) acc[i] = vzero8();

  const float* rs = states + (size_t)st * DS;  // A: one row per lane -> gather free
  const float* re = states + (size_t)en * DS;

  for (int kt = 0; kt < KT_DS - 1; ++kt) {  // seg0: x_start (12 full)
    v16bf a = a_frag_full(rs, kt * 32, half);
#pragma unroll
    for (int nt = 0; nt < NT; ++nt)
      acc[nt] = wmma_bf16(a, b_frag_packed(w1p, (SEG0_BASE + kt) * NT + nt, lane),
                          acc[nt]);
  }
  {
    v16bf a = a_frag_partial(rs, 384, DS - 1, half);
#pragma unroll
    for (int nt = 0; nt < NT; ++nt)
      acc[nt] = wmma_bf16(a, b_frag_packed(w1p, (SEG0_BASE + 12) * NT + nt, lane),
                          acc[nt]);
  }
  for (int kt = 0; kt < KT_DS - 1; ++kt) {  // seg1: x_end (12 full)
    v16bf a = a_frag_full(re, kt * 32, half);
#pragma unroll
    for (int nt = 0; nt < NT; ++nt)
      acc[nt] = wmma_bf16(a, b_frag_packed(w1p, (SEG1_BASE + kt) * NT + nt, lane),
                          acc[nt]);
  }
  {
    v16bf a = a_frag_partial(re, 384, DS - 1, half);
#pragma unroll
    for (int nt = 0; nt < NT; ++nt)
      acc[nt] = wmma_bf16(a, b_frag_packed(w1p, (SEG1_BASE + 12) * NT + nt, lane),
                          acc[nt]);
  }
  for (int kt = 0; kt < KT_DE; ++kt) {  // seg2: pooled (11 full, pad zeroed)
    v16bf a = a_frag_lds(pooledS[wave] + (size_t)n16 * LDP, kt * 32, half);
#pragma unroll
    for (int nt = 0; nt < NT; ++nt)
      acc[nt] = wmma_bf16(a, b_frag_packed(w1p, (SEG2_BASE + kt) * NT + nt, lane),
                          acc[nt]);
  }
  {  // seg3: width embedding (1 partial, clamp to 19; packed B rows >=20 zero)
    const float* rw = wtab + (size_t)bin * DW;
    v16bf a = a_frag_partial(rw, 0, DW - 1, half);
#pragma unroll
    for (int nt = 0; nt < NT; ++nt)
      acc[nt] = wmma_bf16(a, b_frag_packed(w1p, SEG3_BASE * NT + nt, lane), acc[nt]);
  }

  store_h(hS[wave], acc, b1, lane);
  mlp_tail(hS[wave], w2p, b2, w3p, b3, out, spanbase, S_SPAN, lane);
}

// ---------------- workspace layout (bytes, all 256-aligned) ----------------
#define OFF_ATTN 0
#define OFF_PA1 204800                 // 13*10*512*2 = 133120
#define OFF_PA2 (OFF_PA1 + 133120)    // 5*10*512*2 = 51200
#define OFF_PA3 (OFF_PA2 + 51200)     // 5*1*512*2 = 5120
#define OFF_PS1 (OFF_PA3 + 5120)      // 38*10*512*2 = 389120
#define OFF_PS2 (OFF_PS1 + 389120)
#define OFF_PS3 (OFF_PS2 + 51200)

static inline void pack_launch(const float* W, int ldn, int rowoff, int kvalid,
                               int ntiles, int nvalid, int ktiles, __bf16* dst,
                               hipStream_t stream) {
  int total = ktiles * ntiles * 512;
  pack_b_kernel<<<(total + 255) / 256, 256, 0, stream>>>(W, ldn, rowoff, kvalid,
                                                         ntiles, nvalid, total, dst);
}

extern "C" void kernel_launch(void* const* d_in, const int* in_sizes, int n_in,
                              void* d_out, int out_size, void* d_ws, size_t ws_size,
                              hipStream_t stream) {
  const float* states = (const float*)d_in[0];
  const float* embeds = (const float*)d_in[1];
  const int*   sstart = (const int*)d_in[2];
  const int*   swidth = (const int*)d_in[3];
  const float* wtab   = (const float*)d_in[4];
  const float* aw1 = (const float*)d_in[5];
  const float* ab1 = (const float*)d_in[6];
  const float* aw2 = (const float*)d_in[7];
  const float* ab2 = (const float*)d_in[8];
  const float* aw3 = (const float*)d_in[9];
  const float* ab3 = (const float*)d_in[10];
  const float* sw1 = (const float*)d_in[11];
  const float* sb1 = (const float*)d_in[12];
  const float* sw2 = (const float*)d_in[13];
  const float* sb2 = (const float*)d_in[14];
  const float* sw3 = (const float*)d_in[15];
  const float* sb3 = (const float*)d_in[16];

  char* ws = (char*)d_ws;
  float*  attns = (float*)(ws + OFF_ATTN);
  __bf16* pa1 = (__bf16*)(ws + OFF_PA1);
  __bf16* pa2 = (__bf16*)(ws + OFF_PA2);
  __bf16* pa3 = (__bf16*)(ws + OFF_PA3);
  __bf16* ps1 = (__bf16*)(ws + OFF_PS1);
  __bf16* ps2 = (__bf16*)(ws + OFF_PS2);
  __bf16* ps3 = (__bf16*)(ws + OFF_PS3);

  // ---- pack weights to fragment-order padded bf16 (deterministic, every call) ----
  pack_launch(aw1, HH, 0, DS, NT, HH, 13, pa1, stream);
  pack_launch(aw2, HH, 0, HH, NT, HH, 5, pa2, stream);
  pack_launch(aw3, 1, 0, HH, 1, 1, 5, pa3, stream);
  pack_launch(sw1, HH, 0, DS, NT, HH, 13, ps1, stream);                   // x_start
  pack_launch(sw1, HH, DS, DS, NT, HH, 13, ps1 + SEG1_BASE * NT * 512, stream);
  pack_launch(sw1, HH, 2 * DS, DE, NT, HH, 11, ps1 + SEG2_BASE * NT * 512, stream);
  pack_launch(sw1, HH, 2 * DS + DE, DW, NT, HH, 1, ps1 + SEG3_BASE * NT * 512, stream);
  pack_launch(sw2, HH, 0, HH, NT, HH, 5, ps2, stream);
  pack_launch(sw3, 1, 0, HH, 1, 1, 5, ps3, stream);

  int g1 = (T_TOK + 31) / 32;  // 2 waves x 16 rows per block
  attn_logits_kernel<<<g1, 64, 0, stream>>>(states, pa1, ab1, pa2, ab2, pa3, ab3,
                                            attns);

  int g2 = (S_SPAN + 31) / 32;  // 2 waves x 16 spans per block
  span_score_kernel<<<g2, 64, 0, stream>>>(states, embeds, sstart, swidth, wtab, ps1,
                                           sb1, ps2, sb2, ps3, sb3, attns,
                                           (float*)d_out);
}
